// ConvBertAttention_54881092108721
// MI455X (gfx1250) — matrix operand order repaired and checked by reading, not compiled, not run on hardware
//
#include <hip/hip_runtime.h>

// ---------------------------------------------------------------------------
// ConvBertAttention for MI455X (gfx1250): bf16 WMMA GEMM core with LDS
// double-buffering via async global->LDS copies (ASYNCcnt), plus elementwise.
// B=8, L=2048, C=768, H=12, D=64, K=9.  M = B*L = 16384.
// ---------------------------------------------------------------------------

typedef __attribute__((ext_vector_type(16))) __bf16 bf16x16;
typedef __attribute__((ext_vector_type(8)))  float  f32x8;
typedef int i32x4 __attribute__((vector_size(16)));   // matches async-LDS builtin param

#define Cc    768
#define Lseq  2048
#define Hh    12
#define Dd    64
#define Kw    9
#define PADW  4

// GEMM tiling: block 128x128, 8 waves (4 M x 2 N), wave tile 32x64, K step 32
#define TM 128
#define TN 128
#define TK 32
#define LDS_STRIDE 40   // elements; 80B row stride: 16B aligned, conflict-free

// ---- CDNA5 async global->LDS path (guarded: falls back to sync staging) ----
#if defined(__gfx1250__) && __has_builtin(__builtin_amdgcn_global_load_async_to_lds_b128)
#define USE_ASYNC_LDS 1
#else
#define USE_ASYNC_LDS 0
#endif

#define AS1 __attribute__((address_space(1)))
#define AS3 __attribute__((address_space(3)))

__device__ __forceinline__ AS1 i32x4* as_global_v4(const void* p) {
  return (AS1 i32x4*)(unsigned long long)(uintptr_t)p;   // global: identity map
}
__device__ __forceinline__ AS3 i32x4* as_lds_v4(const void* p) {
  return (AS3 i32x4*)(unsigned)(uintptr_t)p;             // LDS: low 32 bits = offset
}

__device__ __forceinline__ void wait_staging() {
#if USE_ASYNC_LDS
#if __has_builtin(__builtin_amdgcn_s_wait_asynccnt)
  __builtin_amdgcn_s_wait_asynccnt(0);
#else
  asm volatile("s_wait_asynccnt 0x0" ::: "memory");
#endif
#endif
}

// ---------------------------------------------------------------------------
// elementwise fp32 -> bf16 convert
__global__ __launch_bounds__(256) void cvt_bf16_kernel(
    const float* __restrict__ x, __bf16* __restrict__ y, size_t n) {
  size_t i = (size_t)blockIdx.x * blockDim.x + threadIdx.x;
  if (i < n) y[i] = (__bf16)x[i];
}

// ---------------------------------------------------------------------------
// depthwise conv over L (kernel 9, zero pad 4), output [M, C] in bf16
__global__ __launch_bounds__(256) void dwconv_kernel(
    const float* __restrict__ X,    // [B, L, C] f32
    const float* __restrict__ dwk,  // [C, 1, 9]
    __bf16* __restrict__ Y)         // [M, C] bf16
{
  size_t idx = (size_t)blockIdx.x * blockDim.x + threadIdx.x; // m*C + c
  int    c   = (int)(idx % Cc);
  size_t m   = idx / Cc;
  int    l   = (int)(m % Lseq);
  size_t b   = m / Lseq;
  float s = 0.0f;
#pragma unroll
  for (int k = 0; k < Kw; ++k) {
    int lp = l + k - PADW;
    if (lp >= 0 && lp < Lseq)
      s += X[(b * Lseq + lp) * Cc + c] * dwk[c * Kw + k];
  }
  Y[idx] = (__bf16)s;
}

// ---------------------------------------------------------------------------
// W [K, N] f32 (row-major) -> Wt [Np, K] bf16, zero-filling rows n >= N
__global__ __launch_bounds__(256) void transpose_w_kernel(
    const float* __restrict__ W, __bf16* __restrict__ Wt,
    int Kdim, int N, int Np)
{
  int k = blockIdx.x * 32 + (threadIdx.x & 31);
  int n = blockIdx.y * 8  + (threadIdx.x >> 5);
  if (k < Kdim && n < Np)
    Wt[(size_t)n * Kdim + k] = (n < N) ? (__bf16)W[(size_t)k * N + n] : (__bf16)0.0f;
}

// ---------------------------------------------------------------------------
// WMMA bf16 GEMM:  Out[M,N] = (A[M,K] * Bt[N,K]^T + bias) [* scale]
// block = 256 threads = 8 waves (4 M x 2 N), block tile 128x128, wave 32x64.
// LDS double-buffered; staging via async global->LDS when available.
template <bool OUT_BF16>
__global__ __launch_bounds__(256) void gemm_bf16_wmma_kernel(
    const __bf16* __restrict__ A,     // [M, Kdim] bf16 row-major
    const __bf16* __restrict__ Bt,    // [N, Kdim] bf16 row-major (B transposed)
    const float*  __restrict__ bias,  // may be null; valid for col < bias_n
    int bias_n,
    const __bf16* __restrict__ scale, // optional elementwise multiplier [M, N]
    void* __restrict__ Out,           // [M, N] f32 or bf16
    int N, int Kdim)
{
  __shared__ __bf16 lA[2][TM * LDS_STRIDE];
  __shared__ __bf16 lB[2][TN * LDS_STRIDE];

  const int m0   = blockIdx.x * TM;
  const int n0   = blockIdx.y * TN;
  const int tid  = threadIdx.x;
  const int lane = tid & 31;
  const int wave = tid >> 5;
  const int wm   = wave >> 1;   // 0..3  (M direction, 32 rows each)
  const int wn   = wave & 1;    // 0..1  (N direction, 64 cols each)
  const int ln   = lane & 15;
  const int lg   = lane >> 4;

  f32x8 acc[2][4] = {};

  // staging coordinates: both tiles are 128 rows x 32 cols -> 2 thr/row, 32B each
  const int srow = tid >> 1, scs = (tid & 1) * 16;

  auto stage = [&](int pp, int k0) {
    const __bf16* gA = A  + (size_t)(m0 + srow) * Kdim + k0 + scs;
    const __bf16* gB = Bt + (size_t)(n0 + srow) * Kdim + k0 + scs;
    __bf16* sA = &lA[pp][srow * LDS_STRIDE + scs];
    __bf16* sB = &lB[pp][srow * LDS_STRIDE + scs];
#if USE_ASYNC_LDS
    __builtin_amdgcn_global_load_async_to_lds_b128(as_global_v4(gA),     as_lds_v4(sA),     0, 0);
    __builtin_amdgcn_global_load_async_to_lds_b128(as_global_v4(gA + 8), as_lds_v4(sA + 8), 0, 0);
    __builtin_amdgcn_global_load_async_to_lds_b128(as_global_v4(gB),     as_lds_v4(sB),     0, 0);
    __builtin_amdgcn_global_load_async_to_lds_b128(as_global_v4(gB + 8), as_lds_v4(sB + 8), 0, 0);
#else
    const uint4* srcA = reinterpret_cast<const uint4*>(gA);
    const uint4* srcB = reinterpret_cast<const uint4*>(gB);
    *reinterpret_cast<uint4*>(sA)     = srcA[0];
    *reinterpret_cast<uint4*>(sA + 8) = srcA[1];
    *reinterpret_cast<uint4*>(sB)     = srcB[0];
    *reinterpret_cast<uint4*>(sB + 8) = srcB[1];
    __builtin_prefetch(gA + TK, 0, 1);
    __builtin_prefetch(gB + TK, 0, 1);
#endif
  };

  stage(0, 0);
  int pp = 0;
  for (int k0 = 0; k0 < Kdim; k0 += TK) {
    wait_staging();       // this wave's async fills of buffer pp complete
    __syncthreads();      // all waves' fills of buffer pp visible
    if (k0 + TK < Kdim) stage(pp ^ 1, k0 + TK);   // overlap next tile fill

    // ---- gather WMMA fragments per ISA 16-bit layouts -------------------
    // A 16x32: lane (m=ln, g=lg): j<8 -> K = g*8 + j ; j>=8 -> K = 16+g*8+(j-8)
    bf16x16 afrag[2];
#pragma unroll
    for (int i = 0; i < 2; ++i) {
      const __bf16* base = &lA[pp][(wm * 32 + i * 16 + ln) * LDS_STRIDE + lg * 8];
      union { bf16x16 v; uint4 q[2]; } u;
      u.q[0] = *reinterpret_cast<const uint4*>(base);       // K = lg*8 .. +7
      u.q[1] = *reinterpret_cast<const uint4*>(base + 16);  // K = 16+lg*8 .. +7
      afrag[i] = u.v;
    }
    // B 32x16: lane (n=ln, g=lg): j<8 -> K = g*8 + j ; j>=8 -> K = 16+g*8+(j-8)
    bf16x16 bfrag[4];
#pragma unroll
    for (int j = 0; j < 4; ++j) {
      const __bf16* base = &lB[pp][(wn * 64 + j * 16 + ln) * LDS_STRIDE + lg * 8];
      union { bf16x16 v; uint4 q[2]; } u;
      u.q[0] = *reinterpret_cast<const uint4*>(base);
      u.q[1] = *reinterpret_cast<const uint4*>(base + 16);
      bfrag[j] = u.v;
    }

    // ---- 8 WMMAs per wave per K step ------------------------------------
#pragma unroll
    for (int i = 0; i < 2; ++i)
#pragma unroll
      for (int j = 0; j < 4; ++j)
        acc[i][j] = __builtin_amdgcn_wmma_f32_16x16x32_bf16(
            false, afrag[i], false, bfrag[j], (short)0, acc[i][j], false, false);

    __syncthreads();      // all waves done reading buffer pp
    pp ^= 1;
  }

  // ---- store D (lane l: N = l%16; VGPR v: M = v + 8*(l/16)) --------------
#pragma unroll
  for (int i = 0; i < 2; ++i) {
#pragma unroll
    for (int j = 0; j < 4; ++j) {
      const int col = n0 + wn * 64 + j * 16 + ln;
      const float bv = (bias && col < bias_n) ? bias[col] : 0.0f;
#pragma unroll
      for (int v = 0; v < 8; ++v) {
        const int row = m0 + wm * 32 + i * 16 + v + 8 * lg;
        float val = acc[i][j][v] + bv;
        if (scale) val *= (float)scale[(size_t)row * N + col];
        if (OUT_BF16)
          ((__bf16*)Out)[(size_t)row * N + col] = (__bf16)val;
        else
          ((float*)Out)[(size_t)row * N + col] = val;
      }
    }
  }
}

// ---------------------------------------------------------------------------
// softmax over kernel dim (9) per (m, h); kern buffer is [M, 128] f32,
// valid cols 0..107 laid out as h*9 + k. In place.
__global__ __launch_bounds__(256) void softmax9_kernel(float* __restrict__ kern, int M) {
  int idx = blockIdx.x * blockDim.x + threadIdx.x; // m*H + h
  if (idx >= M * Hh) return;
  int    h = idx % Hh;
  size_t m = (size_t)(idx / Hh);
  float* p = kern + m * 128 + h * Kw;
  float mx = p[0];
#pragma unroll
  for (int k = 1; k < Kw; ++k) mx = fmaxf(mx, p[k]);
  float s = 0.0f;
  float e[Kw];
#pragma unroll
  for (int k = 0; k < Kw; ++k) { e[k] = __expf(p[k] - mx); s += e[k]; }
  float inv = 1.0f / s;
#pragma unroll
  for (int k = 0; k < Kw; ++k) p[k] = e[k] * inv;
}

// ---------------------------------------------------------------------------
// out[m, c] = sum_k kern[m, h, k] * co[b, l-4+k, c]   (zero-padded in L)
__global__ __launch_bounds__(256) void unfold_mix_kernel(
    const float* __restrict__ co,   // [M, C] f32
    const float* __restrict__ kern, // [M, 128] f32 (h*9+k)
    float* __restrict__ out)        // [M, C] f32
{
  size_t idx = (size_t)blockIdx.x * blockDim.x + threadIdx.x; // m*C + c
  int    c   = (int)(idx % Cc);
  size_t m   = idx / Cc;
  int    h   = c >> 6;
  int    l   = (int)(m % Lseq);
  size_t b   = m / Lseq;
  const float* kp = kern + m * 128 + h * Kw;
  float s = 0.0f;
#pragma unroll
  for (int k = 0; k < Kw; ++k) {
    int lp = l + k - PADW;
    if (lp >= 0 && lp < Lseq)
      s = fmaf(co[(b * Lseq + lp) * Cc + c], kp[k], s);
  }
  out[idx] = s;
}

// ---------------------------------------------------------------------------
extern "C" void kernel_launch(void* const* d_in, const int* in_sizes, int n_in,
                              void* d_out, int out_size, void* d_ws, size_t ws_size,
                              hipStream_t stream) {
  const float* X     = (const float*)d_in[0];  // [B, L, C]
  const float* W_q   = (const float*)d_in[1];  // [C, C]  (in, out)
  const float* b_q   = (const float*)d_in[2];
  const float* dw    = (const float*)d_in[3];  // [C, 1, 9]
  const float* pw    = (const float*)d_in[4];  // [C, C]  (out, in) -> already [N, K]
  const float* sbias = (const float*)d_in[5];
  const float* W_ck  = (const float*)d_in[6];  // [C, 108]
  const float* b_ck  = (const float*)d_in[7];
  const float* W_co  = (const float*)d_in[8];  // [C, C]
  const float* b_co  = (const float*)d_in[9];
  float* out = (float*)d_out;

  const size_t M  = (size_t)in_sizes[0] / Cc;  // 16384
  const size_t MC = M * Cc;

  // ---- workspace layout ----
  char*  base = (char*)d_ws;
  size_t off  = 0;
  auto carve = [&](size_t bytes) -> char* {
    char* p = base + off;
    off = (off + bytes + 255) & ~(size_t)255;
    return p;
  };
  __bf16* Xh   = (__bf16*)carve(MC * 2);             // X in bf16
  __bf16* Dh   = (__bf16*)carve(MC * 2);             // depthwise out, bf16
  __bf16* WqT  = (__bf16*)carve((size_t)Cc * Cc * 2);
  __bf16* WcoT = (__bf16*)carve((size_t)Cc * Cc * 2);
  __bf16* pwN  = (__bf16*)carve((size_t)Cc * Cc * 2);
  __bf16* WckT = (__bf16*)carve((size_t)128 * Cc * 2);
  __bf16* q_h  = (__bf16*)carve(MC * 2);             // q, then conv_attn in place
  float*  co   = (float*) carve(MC * 4);
  float*  kern = (float*) carve(M * 128 * 4);
  (void)ws_size; (void)n_in; (void)out_size;

  // ---- 1. conversions / transposes / depthwise conv ----
  cvt_bf16_kernel<<<dim3((unsigned)(MC / 256)), 256, 0, stream>>>(X, Xh, MC);
  cvt_bf16_kernel<<<dim3((unsigned)((size_t)Cc * Cc / 256)), 256, 0, stream>>>(
      pw, pwN, (size_t)Cc * Cc);
  dwconv_kernel<<<dim3((unsigned)(MC / 256)), 256, 0, stream>>>(X, dw, Dh);
  {
    dim3 g(Cc / 32, Cc / 8);
    transpose_w_kernel<<<g, 256, 0, stream>>>(W_q,  WqT,  Cc, Cc, Cc);
    transpose_w_kernel<<<g, 256, 0, stream>>>(W_co, WcoT, Cc, Cc, Cc);
    dim3 gk(Cc / 32, 128 / 8);
    transpose_w_kernel<<<gk, 256, 0, stream>>>(W_ck, WckT, Cc, Hh * Kw, 128);
  }

  // ---- 2. WMMA GEMMs ----
  dim3 gg((unsigned)(M / TM), Cc / TN);
  gemm_bf16_wmma_kernel<true><<<gg, 256, 0, stream>>>(
      Xh, WqT, b_q, Cc, nullptr, (void*)q_h, Cc, Cc);            // q (bf16)
  gemm_bf16_wmma_kernel<true><<<gg, 256, 0, stream>>>(
      Dh, pwN, sbias, Cc, q_h, (void*)q_h, Cc, Cc);              // conv_attn = key*q
  gemm_bf16_wmma_kernel<false><<<gg, 256, 0, stream>>>(
      Xh, WcoT, b_co, Cc, nullptr, (void*)co, Cc, Cc);           // conv_out (f32)

  dim3 gk2((unsigned)(M / TM), 1);
  gemm_bf16_wmma_kernel<false><<<gk2, 256, 0, stream>>>(
      q_h, WckT, b_ck, Hh * Kw, nullptr, (void*)kern, 128, Cc);  // [M,128] f32

  // ---- 3. softmax + unfold/mix ----
  softmax9_kernel<<<dim3((unsigned)((M * Hh + 255) / 256)), 256, 0, stream>>>(
      kern, (int)M);
  unfold_mix_kernel<<<dim3((unsigned)(MC / 256)), 256, 0, stream>>>(co, kern, out);
}
